// Compressor_58016418234545
// MI455X (gfx1250) — matrix operand order repaired and checked by reading, not compiled, not run on hardware
//
#include <hip/hip_runtime.h>
#include <hip/hip_bf16.h>
#include <math.h>

// Problem constants (match reference)
#define BB   4
#define LL   2048
#define DD   1024
#define NHH  16
#define HDD  64
#define FFF  4096
#define NLL  2
#define BLr  (BB*LL)   // 8192 rows

typedef __attribute__((ext_vector_type(16))) __bf16 v16bf;
typedef __attribute__((ext_vector_type(8)))  float  v8f;

union Frag { v16bf v; unsigned short s[16]; unsigned int u[8]; };
union Acc  { v8f   v; float f[8]; };

__device__ __forceinline__ unsigned short f2bf(float f) {
  unsigned int u = __builtin_bit_cast(unsigned int, f);
  return (unsigned short)((u + 0x7FFFu + ((u >> 16) & 1u)) >> 16);  // RNE
}
__device__ __forceinline__ float bf2f(unsigned short h) {
  unsigned int u = ((unsigned int)h) << 16;
  return __builtin_bit_cast(float, u);
}

// ---------------------------------------------------------------- convert
__global__ void convert_bf16_kernel(unsigned short* __restrict__ dst,
                                    const float* __restrict__ src, size_t n) {
  size_t i = (size_t)blockIdx.x * blockDim.x + threadIdx.x;
  size_t stride = (size_t)gridDim.x * blockDim.x;
  for (; i < n; i += stride) dst[i] = f2bf(src[i]);
}

// ---------------------------------------------------------------- rmsnorm
__global__ __launch_bounds__(256) void rmsnorm_kernel(
    const float* __restrict__ x, const float* __restrict__ w,
    unsigned short* __restrict__ out_bf, float* out_f, int d) {
  int row = blockIdx.x, tid = threadIdx.x;
  const float* xr = x + (size_t)row * d;
  __shared__ float red[256];
  float s = 0.f;
  for (int i = tid; i < d; i += 256) { float v = xr[i]; s += v * v; }
  red[tid] = s; __syncthreads();
  for (int off = 128; off > 0; off >>= 1) {
    if (tid < off) red[tid] += red[tid + off];
    __syncthreads();
  }
  float rs = rsqrtf(red[0] / (float)d + 1e-5f);
  for (int i = tid; i < d; i += 256) {
    float v = xr[i] * rs * w[i];
    out_bf[(size_t)row * d + i] = f2bf(v);
    if (out_f) out_f[(size_t)row * d + i] = v;
  }
}

// ---------------------------------------------------------------- GEMM (bf16 WMMA)
// C[M,N] = A[M,K] * B[K,N]; block tile 256x128, 8 waves as 4x2, wave tile
// 64x64 (16 WMMAs per K-step of 32). A tile staged into LDS with
// GLOBAL_LOAD_ASYNC_TO_LDS_B128 (ASYNCcnt), B tile staged transposed.
// mode 0: Cf=acc   1: Cb=bf16(acc)   2: Cf=acc+resid
// mode 3: Cb=bf16(silu(extra)*acc)   4: Cb=bf16(silu(acc+bias[n]))
__global__ __launch_bounds__(256) void gemm_bf16_kernel(
    const unsigned short* __restrict__ A, const unsigned short* __restrict__ Bm,
    float* Cf, unsigned short* Cb, const float* resid, const float* extra,
    const float* __restrict__ bias, int M, int N, int K, int mode) {
  __shared__ unsigned short sA[256][40];   // [m][k], 80B row (16B multiple)
  __shared__ unsigned short sB[128][34];   // B transposed: [n][k], pad 2
  const int tid = threadIdx.x;
  const int lane = tid & 31, wid = tid >> 5;
  const int hi = lane >> 4, lr = lane & 15;
  const int wm = wid >> 1, wn = wid & 1;           // 4 x 2 wave grid
  const int row0 = blockIdx.y * 256, col0 = blockIdx.x * 128;

  Acc acc[4][4];
  for (int a = 0; a < 4; a++)
    for (int b = 0; b < 4; b++)
      for (int k = 0; k < 8; k++) acc[a][b].f[k] = 0.f;

  const int bk = tid >> 3, bc = (tid & 7) * 16;    // B staging: 16 bf16/thread
  const unsigned int ldsA = (unsigned int)(unsigned long long)&sA[tid][0];

  for (int k0 = 0; k0 < K; k0 += 32) {
    { // stage A row (64B) straight into LDS via async tensor path
      unsigned long long ga =
          (unsigned long long)(A + (size_t)(row0 + tid) * K + k0);
      asm volatile(
          "global_load_async_to_lds_b128 %0, %1, off\n\t"
          "global_load_async_to_lds_b128 %0, %1, off offset:16\n\t"
          "global_load_async_to_lds_b128 %0, %1, off offset:32\n\t"
          "global_load_async_to_lds_b128 %0, %1, off offset:48"
          :: "v"(ldsA), "v"(ga) : "memory");
    }
    { // stage B tile transposed (needed for B-fragment layout)
      const unsigned short* src = Bm + (size_t)(k0 + bk) * N + col0 + bc;
      uint4 p0 = *(const uint4*)src;
      uint4 p1 = *(const uint4*)(src + 8);
      unsigned short tmp[16];
      *(uint4*)&tmp[0] = p0; *(uint4*)&tmp[8] = p1;
#pragma unroll
      for (int i = 0; i < 16; i++) sB[bc + i][bk] = tmp[i];
    }
    if (k0 + 32 < K)  // global_prefetch_b8 of next B K-tile
      __builtin_prefetch(Bm + (size_t)(k0 + 32 + bk) * N + col0 + bc, 0, 3);
    asm volatile("s_wait_asynccnt 0" ::: "memory");
    __syncthreads();

    Frag af[4], bfr[4];
#pragma unroll
    for (int tm = 0; tm < 4; tm++) {
      int r = wm * 64 + tm * 16 + lr;
#pragma unroll
      for (int v = 0; v < 8; v++) {
        int kk = ((v >> 2) << 4) + (hi << 3) + ((v & 3) << 1);  // 16-bit A layout
        af[tm].u[v] = *(const unsigned int*)&sA[r][kk];
      }
    }
#pragma unroll
    for (int tn = 0; tn < 4; tn++) {
      int c = wn * 64 + tn * 16 + lr;
#pragma unroll
      for (int v = 0; v < 8; v++) {
        int kk = ((v >> 2) << 4) + (hi << 3) + ((v & 3) << 1);  // 16-bit B layout
        bfr[tn].u[v] = *(const unsigned int*)&sB[c][kk];
      }
    }
#pragma unroll
    for (int tm = 0; tm < 4; tm++)
#pragma unroll
      for (int tn = 0; tn < 4; tn++)
        acc[tm][tn].v = __builtin_amdgcn_wmma_f32_16x16x32_bf16(
            false, af[tm].v, false, bfr[tn].v, (short)0, acc[tm][tn].v, false, false);
    __syncthreads();
  }

  // epilogue (C layout: VGPR r -> row r + 8*hi, col = lr)
#pragma unroll
  for (int tm = 0; tm < 4; tm++)
#pragma unroll
    for (int tn = 0; tn < 4; tn++)
#pragma unroll
      for (int r = 0; r < 8; r++) {
        int m = row0 + wm * 64 + tm * 16 + r + (hi << 3);
        int n = col0 + wn * 64 + tn * 16 + lr;
        size_t idx = (size_t)m * N + n;
        float val = acc[tm][tn].f[r];
        if (mode == 0)      { Cf[idx] = val; }
        else if (mode == 1) { Cb[idx] = f2bf(val); }
        else if (mode == 2) { Cf[idx] = val + resid[idx]; }
        else if (mode == 3) {
          float a = extra[idx];
          float sg = a / (1.f + __expf(-a));
          Cb[idx] = f2bf(sg * val);
        } else {
          float t = val + bias[n];
          Cb[idx] = f2bf(t / (1.f + __expf(-t)));
        }
      }
}

// ---------------------------------------------------------------- RoPE (in place)
__global__ void rope_kernel(unsigned short* t, const float* __restrict__ cosp,
                            const float* __restrict__ sinp) {
  size_t i = (size_t)blockIdx.x * blockDim.x + threadIdx.x;
  const size_t total = (size_t)BB * LL * NHH * 32;
  if (i >= total) return;
  int d = (int)(i & 31);
  size_t r = i >> 5;
  int h = (int)(r % NHH); r /= NHH;
  int l = (int)(r % LL);
  int b = (int)(r / LL);
  size_t base = ((size_t)(b * LL + l) * NHH + h) * HDD;
  float x1 = bf2f(t[base + d]), x2 = bf2f(t[base + d + 32]);
  float c1 = cosp[l * HDD + d],      s1 = sinp[l * HDD + d];
  float c2 = cosp[l * HDD + d + 32], s2 = sinp[l * HDD + d + 32];
  t[base + d]      = f2bf(x1 * c1 - x2 * s1);
  t[base + d + 32] = f2bf(x2 * c2 + x1 * s2);
}

// ---------------------------------------------------------------- flash attention
// grid (L/64, NH, B), 128 threads = 4 waves; wave owns 16 query rows.
__global__ __launch_bounds__(128) void flash_attn_kernel(
    const unsigned short* __restrict__ Q, const unsigned short* __restrict__ Kk,
    const unsigned short* __restrict__ V, unsigned short* __restrict__ O) {
  const int qt = blockIdx.x, h = blockIdx.y, b = blockIdx.z;
  const int tid = threadIdx.x, lane = tid & 31, wid = tid >> 5;
  const int hi = lane >> 4, lr = lane & 15;
  const int qbase = qt * 64 + wid * 16;
  const int RS = NHH * HDD;  // row stride = D
  const float scale = 0.125f;  // 1/sqrt(64)
  __shared__ unsigned short sP[4][16][66];  // per-wave P tile

  Frag qf[2];
#pragma unroll
  for (int kb = 0; kb < 2; kb++)
#pragma unroll
    for (int v = 0; v < 8; v++) {
      int kk = kb * 32 + ((v >> 2) << 4) + (hi << 3) + ((v & 3) << 1);
      size_t off = (size_t)(b * LL + qbase + lr) * RS + h * HDD + kk;
      qf[kb].u[v] = *(const unsigned int*)(Q + off);
    }

  float mrun[8], lrun[8];
#pragma unroll
  for (int r = 0; r < 8; r++) { mrun[r] = -1e30f; lrun[r] = 0.f; }
  Acc oacc[4];
  for (int tn = 0; tn < 4; tn++)
    for (int r = 0; r < 8; r++) oacc[tn].f[r] = 0.f;

  for (int j = 0; j <= qt; j++) {
    const int kb0 = j * 64;
    Acc sacc[4];
#pragma unroll
    for (int tn = 0; tn < 4; tn++) {
      for (int r = 0; r < 8; r++) sacc[tn].f[r] = 0.f;
#pragma unroll
      for (int kb = 0; kb < 2; kb++) {
        Frag kf;  // B fragment of K^T: rows = head dims, cols = keys
#pragma unroll
        for (int v = 0; v < 8; v++) {
          int kd = kb * 32 + ((v >> 2) << 4) + (hi << 3) + ((v & 3) << 1);
          int key = kb0 + tn * 16 + lr;
          size_t off = (size_t)(b * LL + key) * RS + h * HDD + kd;
          kf.u[v] = *(const unsigned int*)(Kk + off);
        }
        sacc[tn].v = __builtin_amdgcn_wmma_f32_16x16x32_bf16(
            false, qf[kb].v, false, kf.v, (short)0, sacc[tn].v, false, false);
      }
    }
    const bool diag = (j == qt);
#pragma unroll
    for (int r = 0; r < 8; r++) {
      int qrow = qbase + r + (hi << 3);
      float mx = mrun[r];
#pragma unroll
      for (int tn = 0; tn < 4; tn++) {
        float sv = sacc[tn].f[r] * scale;
        if (diag && (kb0 + tn * 16 + lr) > qrow) sv = -1e30f;
        sacc[tn].f[r] = sv;
        mx = fmaxf(mx, sv);
      }
#pragma unroll
      for (int off = 8; off >= 1; off >>= 1) mx = fmaxf(mx, __shfl_xor(mx, off, 32));
      float corr = __expf(mrun[r] - mx);
      lrun[r] *= corr;
#pragma unroll
      for (int tn = 0; tn < 4; tn++) oacc[tn].f[r] *= corr;
      float rs = 0.f;
#pragma unroll
      for (int tn = 0; tn < 4; tn++) {
        float p = __expf(sacc[tn].f[r] - mx);
        sacc[tn].f[r] = p; rs += p;
      }
#pragma unroll
      for (int off = 8; off >= 1; off >>= 1) rs += __shfl_xor(rs, off, 32);
      lrun[r] += rs; mrun[r] = mx;
    }
    // P (C-layout) -> LDS -> A-fragment layout
#pragma unroll
    for (int tn = 0; tn < 4; tn++)
#pragma unroll
      for (int r = 0; r < 8; r++)
        sP[wid][r + (hi << 3)][tn * 16 + lr] = f2bf(sacc[tn].f[r]);
    asm volatile("s_wait_dscnt 0" ::: "memory");
    Frag pf[2];
#pragma unroll
    for (int kb = 0; kb < 2; kb++)
#pragma unroll
      for (int v = 0; v < 8; v++) {
        int kk = kb * 32 + ((v >> 2) << 4) + (hi << 3) + ((v & 3) << 1);
        pf[kb].u[v] = *(const unsigned int*)&sP[wid][lr][kk];
      }
#pragma unroll
    for (int kb = 0; kb < 2; kb++)
#pragma unroll
      for (int tn = 0; tn < 4; tn++) {
        Frag vf;  // B fragment of V: rows = keys, cols = head dims
#pragma unroll
        for (int v = 0; v < 8; v++) {
          int key = kb0 + kb * 32 + ((v >> 2) << 4) + (hi << 3) + ((v & 3) << 1);
          size_t off0 = (size_t)(b * LL + key) * RS + h * HDD + tn * 16 + lr;
          unsigned int lo = V[off0];
          unsigned int hv = V[off0 + RS];
          vf.u[v] = lo | (hv << 16);
        }
        oacc[tn].v = __builtin_amdgcn_wmma_f32_16x16x32_bf16(
            false, pf[kb].v, false, vf.v, (short)0, oacc[tn].v, false, false);
      }
  }
#pragma unroll
  for (int tn = 0; tn < 4; tn++)
#pragma unroll
    for (int r = 0; r < 8; r++) {
      int qrow = qbase + r + (hi << 3);
      float val = oacc[tn].f[r] / lrun[r];
      O[(size_t)(b * LL + qrow) * RS + h * HDD + tn * 16 + lr] = f2bf(val);
    }
}

// ---------------------------------------------------------------- router tail
__global__ __launch_bounds__(256) void router_kernel(
    const unsigned short* __restrict__ r1, const float* __restrict__ w2,
    const float* __restrict__ b2, float* __restrict__ maskf) {
  int row = blockIdx.x, tid = threadIdx.x;
  __shared__ float red[256];
  float s = 0.f;
  for (int i = tid; i < DD; i += 256) s += bf2f(r1[(size_t)row * DD + i]) * w2[i];
  red[tid] = s; __syncthreads();
  for (int off = 128; off > 0; off >>= 1) {
    if (tid < off) red[tid] += red[tid + off];
    __syncthreads();
  }
  if (tid == 0) {
    float logit = red[0] + b2[0];
    float p = 1.f / (1.f + __expf(-logit));
    maskf[row] = (p > 0.5f) ? 1.f : 0.f;
  }
}

__global__ __launch_bounds__(256) void scan_kernel(
    const float* __restrict__ maskf, int* __restrict__ bpos, int* __restrict__ counts,
    float* __restrict__ bound_out, float* __restrict__ counts_out) {
  int b = blockIdx.x, tid = threadIdx.x;
  __shared__ int pre[256];
  __shared__ int total_s;
  const int CH = LL / 256;  // 8
  int local = 0;
  for (int k = 0; k < CH; k++) {
    int l = tid * CH + k;
    float mv = (l == LL - 1) ? 1.f : maskf[b * LL + l];
    local += (mv > 0.5f) ? 1 : 0;
  }
  pre[tid] = local; __syncthreads();
  if (tid == 0) {
    int run = 0;
    for (int i = 0; i < 256; i++) { int t = pre[i]; pre[i] = run; run += t; }
    total_s = run;
  }
  __syncthreads();
  int run = pre[tid];
  int zrun = tid * CH - pre[tid];
  int tot = total_s;
  for (int k = 0; k < CH; k++) {
    int l = tid * CH + k;
    float mv = (l == LL - 1) ? 1.f : maskf[b * LL + l];
    if (mv > 0.5f) {
      bpos[b * LL + run] = l;
      bound_out[(size_t)b * LL + run] = (float)l;
      run++;
    } else {
      bound_out[(size_t)b * LL + tot + zrun] = (float)l;
      zrun++;
    }
  }
  if (tid == 0) { counts[b] = tot; counts_out[b] = (float)tot; }
}

__global__ __launch_bounds__(256) void gather_kernel(
    float* __restrict__ comp, const float* __restrict__ xf,
    const int* __restrict__ bpos, const int* __restrict__ counts) {
  int bs = blockIdx.x;
  int b = bs / LL, s = bs % LL, tid = threadIdx.x;
  float* dst = comp + (size_t)bs * DD;
  if (s < counts[b]) {
    const float* src = xf + (size_t)(b * LL + bpos[b * LL + s]) * DD;
    for (int i = tid; i < DD; i += 256) dst[i] = src[i];
  } else {
    for (int i = tid; i < DD; i += 256) dst[i] = 0.f;
  }
}

__global__ void avg_kernel(const int* __restrict__ counts, float* __restrict__ avg_out) {
  if (threadIdx.x == 0 && blockIdx.x == 0) {
    int sum = 0;
    for (int b = 0; b < BB; b++) sum += counts[b];
    avg_out[0] = (float)LL * (float)BB / (float)sum;
  }
}

// ================================================================ host
extern "C" void kernel_launch(void* const* d_in, const int* in_sizes, int n_in,
                              void* d_out, int out_size, void* d_ws, size_t ws_size,
                              hipStream_t stream) {
  (void)in_sizes; (void)n_in; (void)out_size; (void)ws_size;
  const float* x_in  = (const float*)d_in[0];
  const float* cosp  = (const float*)d_in[1];
  const float* sinp  = (const float*)d_in[2];
  const float* ln1w  = (const float*)d_in[3];
  const float* ln2w  = (const float*)d_in[4];
  const float* wq    = (const float*)d_in[5];
  const float* wk    = (const float*)d_in[6];
  const float* wv    = (const float*)d_in[7];
  const float* wo    = (const float*)d_in[8];
  const float* w1    = (const float*)d_in[9];
  const float* w3    = (const float*)d_in[10];
  const float* w2    = (const float*)d_in[11];
  const float* normw = (const float*)d_in[12];
  const float* mw1   = (const float*)d_in[13];
  const float* mb1   = (const float*)d_in[14];
  const float* mw2   = (const float*)d_in[15];
  const float* mb2   = (const float*)d_in[16];

  char* ws = (char*)d_ws;
  size_t cur = 0;
  auto alloc = [&](size_t bytes) -> char* {
    char* p = ws + cur;
    cur += (bytes + 255) & ~(size_t)255;
    return p;
  };
  float*          x_cur  = (float*)alloc((size_t)BLr * DD * 4);
  unsigned short* h_bf   = (unsigned short*)alloc((size_t)BLr * DD * 2);
  unsigned short* q_bf   = (unsigned short*)alloc((size_t)BLr * DD * 2);
  unsigned short* k_bf   = (unsigned short*)alloc((size_t)BLr * DD * 2);
  unsigned short* v_bf   = (unsigned short*)alloc((size_t)BLr * DD * 2);
  unsigned short* o_bf   = (unsigned short*)alloc((size_t)BLr * DD * 2);
  float*          a_f32  = (float*)alloc((size_t)BLr * FFF * 4);
  unsigned short* g_bf   = (unsigned short*)alloc((size_t)BLr * FFF * 2);
  unsigned short* wq_bf  = (unsigned short*)alloc((size_t)DD * DD * 2);
  unsigned short* wk_bf  = (unsigned short*)alloc((size_t)DD * DD * 2);
  unsigned short* wv_bf  = (unsigned short*)alloc((size_t)DD * DD * 2);
  unsigned short* wo_bf  = (unsigned short*)alloc((size_t)DD * DD * 2);
  unsigned short* w1_bf  = (unsigned short*)alloc((size_t)DD * FFF * 2);
  unsigned short* w3_bf  = (unsigned short*)alloc((size_t)DD * FFF * 2);
  unsigned short* w2_bf  = (unsigned short*)alloc((size_t)FFF * DD * 2);
  unsigned short* mw1_bf = (unsigned short*)alloc((size_t)DD * DD * 2);
  unsigned short* r1_bf  = (unsigned short*)alloc((size_t)BLr * DD * 2);
  float*          maskf  = (float*)alloc((size_t)BLr * 4);
  int*            bpos   = (int*)alloc((size_t)BLr * 4);
  int*            cnts   = (int*)alloc((size_t)BB * 4);

  auto conv = [&](unsigned short* dst, const float* src, size_t n) {
    convert_bf16_kernel<<<1024, 256, 0, stream>>>(dst, src, n);
  };
  auto gemm = [&](const unsigned short* A, const unsigned short* Bm, float* Cf,
                  unsigned short* Cb, const float* resid, const float* extra,
                  const float* bias, int M, int N, int K, int mode) {
    dim3 g(N / 128, M / 256);
    gemm_bf16_kernel<<<g, 256, 0, stream>>>(A, Bm, Cf, Cb, resid, extra, bias, M, N, K, mode);
  };

  hipMemcpyAsync(x_cur, x_in, (size_t)BLr * DD * 4, hipMemcpyDeviceToDevice, stream);

  for (int i = 0; i < NLL; i++) {
    conv(wq_bf, wq + (size_t)i * DD * DD, (size_t)DD * DD);
    conv(wk_bf, wk + (size_t)i * DD * DD, (size_t)DD * DD);
    conv(wv_bf, wv + (size_t)i * DD * DD, (size_t)DD * DD);
    conv(wo_bf, wo + (size_t)i * DD * DD, (size_t)DD * DD);
    conv(w1_bf, w1 + (size_t)i * DD * FFF, (size_t)DD * FFF);
    conv(w3_bf, w3 + (size_t)i * DD * FFF, (size_t)DD * FFF);
    conv(w2_bf, w2 + (size_t)i * FFF * DD, (size_t)FFF * DD);

    rmsnorm_kernel<<<BLr, 256, 0, stream>>>(x_cur, ln1w + (size_t)i * DD, h_bf, nullptr, DD);
    gemm(h_bf, wq_bf, nullptr, q_bf, nullptr, nullptr, nullptr, BLr, DD, DD, 1);
    gemm(h_bf, wk_bf, nullptr, k_bf, nullptr, nullptr, nullptr, BLr, DD, DD, 1);
    gemm(h_bf, wv_bf, nullptr, v_bf, nullptr, nullptr, nullptr, BLr, DD, DD, 1);
    {
      int nthr = BB * LL * NHH * 32;
      rope_kernel<<<nthr / 256, 256, 0, stream>>>(q_bf, cosp, sinp);
      rope_kernel<<<nthr / 256, 256, 0, stream>>>(k_bf, cosp, sinp);
    }
    flash_attn_kernel<<<dim3(LL / 64, NHH, BB), 128, 0, stream>>>(q_bf, k_bf, v_bf, o_bf);
    gemm(o_bf, wo_bf, x_cur, nullptr, x_cur, nullptr, nullptr, BLr, DD, DD, 2);

    rmsnorm_kernel<<<BLr, 256, 0, stream>>>(x_cur, ln2w + (size_t)i * DD, h_bf, nullptr, DD);
    gemm(h_bf, w1_bf, a_f32, nullptr, nullptr, nullptr, nullptr, BLr, FFF, DD, 0);
    gemm(h_bf, w3_bf, nullptr, g_bf, nullptr, a_f32, nullptr, BLr, FFF, DD, 3);
    gemm(g_bf, w2_bf, x_cur, nullptr, x_cur, nullptr, nullptr, BLr, DD, FFF, 2);
  }

  float* x_out      = (float*)d_out;
  float* comp_out   = x_out + (size_t)BLr * DD;
  float* bound_out  = comp_out + (size_t)BLr * DD;
  float* counts_out = bound_out + (size_t)BB * LL;
  float* avg_out    = counts_out + BB;

  rmsnorm_kernel<<<BLr, 256, 0, stream>>>(x_cur, normw, h_bf, x_out, DD);
  conv(mw1_bf, mw1, (size_t)DD * DD);
  gemm(h_bf, mw1_bf, nullptr, r1_bf, nullptr, nullptr, mb1, BLr, DD, DD, 4);
  router_kernel<<<BLr, 256, 0, stream>>>(r1_bf, mw2, mb2, maskf);
  scan_kernel<<<BB, 256, 0, stream>>>(maskf, bpos, cnts, bound_out, counts_out);
  gather_kernel<<<BB * LL, 256, 0, stream>>>(comp_out, x_out, bpos, cnts);
  avg_kernel<<<1, 32, 0, stream>>>(cnts, avg_out);
}
